// DRNN_18665927868848
// MI455X (gfx1250) — matrix operand order repaired and checked by reading, not compiled, not run on hardware
//
#include <hip/hip_runtime.h>
#include <hip/hip_bf16.h>

#define HIDDEN  1024
#define BATCH   128
#define TSTEPS  512
#define NBLOCKS 32              // 1 per WGP; each owns 32 hidden units (128 gate rows)
#define THREADS 256             // 8 wave32
#define KSTEPS  (HIDDEN / 32)   // 32 WMMA K-steps
#define LDS_BYTES (8 * KSTEPS * 32 * 32)  // 8 N-tiles * 32 K-steps * 32 lanes * 32 B = 256 KB

typedef __attribute__((ext_vector_type(16))) __bf16 v16bf;
typedef __attribute__((ext_vector_type(8)))  __bf16 v8bf;
typedef __attribute__((ext_vector_type(8)))  float  v8f;

__device__ __forceinline__ float sigf(float x)  { return 1.0f / (1.0f + __expf(-x)); }
__device__ __forceinline__ float tanhf_(float x){ return 1.0f - 2.0f / (__expf(2.0f * x) + 1.0f); }

__device__ __forceinline__ void grid_sync(int* bar) {
    __syncthreads();
    __threadfence();
    if (threadIdx.x == 0) {
        volatile int* vgen = bar + 1;
        int g = *vgen;
        if (atomicAdd(bar, 1) == NBLOCKS - 1) {
            atomicExch(bar, 0);
            __threadfence();
            atomicAdd(bar + 1, 1);
        } else {
            while (*vgen == g) { __builtin_amdgcn_s_sleep(2); }
        }
    }
    __syncthreads();
    __threadfence();
}

// Zero h ping-pong buffers, seed out with b_lin, reset the grid barrier.
__global__ void lstm_init(float* out, const float* b_lin, unsigned int* hzero, int* bar) {
    size_t i = (size_t)blockIdx.x * blockDim.x + threadIdx.x;
    size_t stride = (size_t)gridDim.x * blockDim.x;
    const size_t HDW = (size_t)2 * BATCH * HIDDEN / 2;  // bf16 dwords
    for (size_t j = i; j < HDW; j += stride) hzero[j] = 0u;
    float bl = b_lin[0];
    for (size_t j = i; j < (size_t)BATCH * TSTEPS; j += stride) out[j] = bl;
    if (i == 0) { bar[0] = 0; bar[1] = 0; }
}

__global__ __launch_bounds__(THREADS)
void lstm_persistent(const float* __restrict__ x,      // [B,T]
                     const float* __restrict__ W_ih,   // [4H]
                     const float* __restrict__ W_hh,   // [4H,H]
                     const float* __restrict__ b_ih,   // [4H]
                     const float* __restrict__ b_hh,   // [4H]
                     const float* __restrict__ W_lin,  // [H]
                     float* __restrict__ out,          // [B,T] (pre-seeded with b_lin)
                     __bf16* __restrict__ hbuf,        // [2][B][H] bf16 ping-pong
                     int* __restrict__ bar) {
    extern __shared__ __align__(32) char smem_raw[];
    __bf16* bsm = (__bf16*)smem_raw;

    const int tid = threadIdx.x;
    const int bk  = blockIdx.x;      // hidden-unit slice: units [32*bk, 32*bk+32)
    const int w   = tid >> 5;        // wave id: owns batch rows [16w, 16w+16)
    const int l   = tid & 31;
    const int hl  = l >> 4;          // lane half (WMMA layout)
    const int ln  = l & 15;

    // ---- one-time: pack this block's W_hh slice into LDS as bf16 B-fragments ----
    // local gate row n in [0,128): n = 32*gate + unit -> global row r = gate*1024 + 32*bk + unit
    // B fragment layout (16-bit, 32x16 tile): lane = 16*((k>>3)&1) + (n&15), elem = (k&7) + 8*((k>>4)&1)
    for (int idx = tid; idx < 128 * HIDDEN; idx += THREADS) {
        int n = idx >> 10;
        int k = idx & (HIDDEN - 1);
        int r = ((n >> 5) << 10) + (bk << 5) + (n & 31);
        float wv = W_hh[(size_t)r * HIDDEN + k];
        int frag = (((n >> 4) << 5) + (k >> 5)) * 32 + ((((k >> 3) & 1) << 4) + (n & 15));
        bsm[frag * 16 + ((k & 7) + (((k >> 4) & 1) << 3))] = (__bf16)wv;
    }
    __syncthreads();

    // per-lane constants: for N-tile nt this lane holds column n = 16*nt + ln
    float wih_f[8], bias_f[8];
#pragma unroll
    for (int nt = 0; nt < 8; ++nt) {
        int u = ((nt & 1) << 4) + ln;
        int r = ((nt >> 1) << 10) + (bk << 5) + u;
        wih_f[nt]  = W_ih[r];
        bias_f[nt] = b_ih[r] + b_hh[r];
    }
    const float wl0 = W_lin[(bk << 5) + ln];
    const float wl1 = W_lin[(bk << 5) + ln + 16];

    const int b0 = (w << 4) + (hl << 3);   // acc VGPR v covers batch b0+v for this lane
    float cst[16];                          // c state: [v][uu], resident across all steps
#pragma unroll
    for (int i = 0; i < 16; ++i) cst[i] = 0.0f;

    const v16bf* bfr = (const v16bf*)smem_raw;

    for (int t = 0; t < TSTEPS; ++t) {
        const __bf16* hin  = hbuf + (size_t)(t & 1) * (BATCH * HIDDEN);
        __bf16*       hout = hbuf + (size_t)((t + 1) & 1) * (BATCH * HIDDEN);

        float xv[8];
#pragma unroll
        for (int v = 0; v < 8; ++v) xv[v] = x[(size_t)(b0 + v) * TSTEPS + t];

        // acc init: x_t * W_ih + (b_ih + b_hh)
        v8f acc[8];
#pragma unroll
        for (int nt = 0; nt < 8; ++nt) {
#pragma unroll
            for (int v = 0; v < 8; ++v) acc[nt][v] = xv[v] * wih_f[nt] + bias_f[nt];
        }

        // GEMM: [16 x 128] += [16 x 1024]_bf16 * [1024 x 128]_bf16
        // A layout (16-bit 16x32): lane row m = ln; elems 0..7 -> k = 8*hl..+7; 8..15 -> +16
        const __bf16* hrow = hin + (size_t)((w << 4) + ln) * HIDDEN + (hl << 3);
        for (int ks = 0; ks < KSTEPS; ++ks) {
            v8bf lo = *(const v8bf*)(hrow);
            v8bf hi = *(const v8bf*)(hrow + 16);
            v16bf a = __builtin_shufflevector(lo, hi, 0, 1, 2, 3, 4, 5, 6, 7,
                                                      8, 9, 10, 11, 12, 13, 14, 15);
            hrow += 32;
#pragma unroll
            for (int nt = 0; nt < 8; ++nt) {
                v16bf bm = bfr[((nt << 5) + ks) * 32 + l];
                acc[nt] = __builtin_amdgcn_wmma_f32_16x16x32_bf16(
                    false, a, false, bm, (short)0, acc[nt], false, false);
            }
        }

        // LSTM cell update + linear-head partial (all in registers)
#pragma unroll
        for (int v = 0; v < 8; ++v) {
            int b = b0 + v;
            float s = 0.0f;
#pragma unroll
            for (int uu = 0; uu < 2; ++uu) {
                float ig = acc[0 + uu][v];
                float fg = acc[2 + uu][v];
                float gg = acc[4 + uu][v];
                float og = acc[6 + uu][v];
                float c  = cst[(v << 1) + uu];
                float cn = sigf(fg) * c + sigf(ig) * tanhf_(gg);
                float hn = sigf(og) * tanhf_(cn);
                cst[(v << 1) + uu] = cn;
                int u = (uu << 4) + ln;
                hout[(size_t)b * HIDDEN + (bk << 5) + u] = (__bf16)hn;
                s += hn * (uu ? wl1 : wl0);
            }
            // reduce over the 16 lanes sharing batch row b (xor over lane bits 0..3)
            s += __shfl_xor(s, 1, 32);
            s += __shfl_xor(s, 2, 32);
            s += __shfl_xor(s, 4, 32);
            s += __shfl_xor(s, 8, 32);
            if (ln == 0) atomicAdd(out + (size_t)b * TSTEPS + t, s);
        }

        grid_sync(bar);   // h(t+1) visible device-wide before next step
    }
}

extern "C" void kernel_launch(void* const* d_in, const int* in_sizes, int n_in,
                              void* d_out, int out_size, void* d_ws, size_t ws_size,
                              hipStream_t stream) {
    const float* x     = (const float*)d_in[0];
    const float* W_ih  = (const float*)d_in[1];
    const float* W_hh  = (const float*)d_in[2];
    const float* b_ih  = (const float*)d_in[3];
    const float* b_hh  = (const float*)d_in[4];
    const float* W_lin = (const float*)d_in[5];
    const float* b_lin = (const float*)d_in[6];
    float* out = (float*)d_out;

    __bf16* hbuf = (__bf16*)d_ws;                                   // 512 KB ping-pong
    int* bar = (int*)((char*)d_ws + (size_t)2 * BATCH * HIDDEN * sizeof(__bf16));

    lstm_init<<<256, 256, 0, stream>>>(out, b_lin, (unsigned int*)d_ws, bar);
    lstm_persistent<<<NBLOCKS, THREADS, LDS_BYTES, stream>>>(
        x, W_ih, W_hh, b_ih, b_hh, W_lin, out, hbuf, bar);
}
